// MessagePassingEdgeModel_5420248727648
// MI455X (gfx1250) — matrix operand order repaired and checked by reading, not compiled, not run on hardware
//
#include <hip/hip_runtime.h>

// ---------------------------------------------------------------------------
// MessagePassingEdgeModel for MI455X (gfx1250, wave32, WMMA)
//
// h = concat(x_s[src], x_t[tgt], edge_attr, x_u)      [E, 256]
// h = LeakyReLU(h @ W1 + b1)                          [E, 256]
// h = h @ W2 + b2                                     [E, 64]
// out = RMSNorm(h) * gamma                            [E, 64]  (f32)
//
// f16 WMMA (16x16x32, f32 accum). Weights pre-packed (one-time kernel) into a
// FRAGMENT-PACKED f16 layout in d_ws -> LDS: each (kb,nb) 16x32 B-tile is
// 1KB contiguous, lane stride 16B => conflict-free LDS banking. H staging
// uses the analogous A-fragment-packed layout. B fragments are pipelined
// TWO tiles deep so ds_load latency hides behind two WMMAs; gather loads are
// software-pipelined 1 K-block deep with sched_barrier fencing.
// ---------------------------------------------------------------------------

typedef _Float16 h16 __attribute__((ext_vector_type(16)));
typedef _Float16 h8  __attribute__((ext_vector_type(8)));
typedef __fp16   p2  __attribute__((ext_vector_type(2)));   // cvt_pkrtz result
typedef float    f4  __attribute__((ext_vector_type(4)));
typedef float    v8f __attribute__((ext_vector_type(8)));

#define DGLOB  64
#define MSG    256
#define LEAKY  0.01f
#define RMSEPS 1.1920928955078125e-07f   // finfo(float32).eps

// LDS layout (halves):
//   sW1 : 8 kb * 16 nb * 512 = 65536   (131072 B)
//   sW2 : 8 kb *  4 nb * 512 = 16384   ( 32768 B)
//   sH  : 8 waves * 8 kb * 512 = 32768 ( 65536 B)
//   sB1 : 256 f32 ; sB2 : 64 f32 ; sG : 64 f32
#define SMEM_BYTES (131072 + 32768 + 65536 + 1024 + 256 + 256)

// -------------------------- helpers ----------------------------------------

// pack 8 f32 (two f4 regs) -> 8 f16
static __device__ inline h8 pk8(f4 a, f4 b) {
  p2 t0 = __builtin_amdgcn_cvt_pkrtz(a[0], a[1]);
  p2 t1 = __builtin_amdgcn_cvt_pkrtz(a[2], a[3]);
  p2 t2 = __builtin_amdgcn_cvt_pkrtz(b[0], b[1]);
  p2 t3 = __builtin_amdgcn_cvt_pkrtz(b[2], b[3]);
  h8 r;
  r[0] = (_Float16)t0[0]; r[1] = (_Float16)t0[1];
  r[2] = (_Float16)t1[0]; r[3] = (_Float16)t1[1];
  r[4] = (_Float16)t2[0]; r[5] = (_Float16)t2[1];
  r[6] = (_Float16)t3[0]; r[7] = (_Float16)t3[1];
  return r;
}

static __device__ inline h16 cat16(h8 a, h8 b) {
  h16 r;
#pragma unroll
  for (int i = 0; i < 8; ++i) { r[i] = a[i]; r[i + 8] = b[i]; }
  return r;
}

// fragment tile read: piece0 at p, piece1 at p+256 halves (both 16B/lane)
static __device__ inline h16 ldfrag(const _Float16* p) {
  h8 a = *(const h8*)p;
  h8 b = *(const h8*)(p + 256);
  return cat16(a, b);
}

// -------------------------- kernel 1: weight pack ---------------------------
// Fragment-packed f16 layout in d_ws:
//  W1: tid = ((kb*16+nb)*512) + piece*256 + lane*8 + j       (65536 elems)
//      value = W1[k*256+n], n = nb*16+(lane&15),
//              k = kb*32 + piece*16 + ((lane>>4)&1)*8 + j
//  W2: same with 4 nb tiles, value = W2[k*64+n]              (16384 elems)
__global__ __launch_bounds__(256) void mp_pack_w(
    const float* __restrict__ W1, const float* __restrict__ W2,
    _Float16* __restrict__ wsW) {
  int tid = blockIdx.x * 256 + threadIdx.x;
  if (tid < 65536) {
    int kb = tid >> 13;
    int nb = (tid >> 9) & 15;
    int pc = (tid >> 8) & 1;
    int l  = (tid >> 3) & 31;
    int j  = tid & 7;
    int n = nb * 16 + (l & 15);
    int k = kb * 32 + pc * 16 + ((l >> 4) & 1) * 8 + j;
    wsW[tid] = (_Float16)W1[k * MSG + n];
  } else if (tid < 65536 + 16384) {
    int t2 = tid - 65536;
    int kb = t2 >> 11;
    int nb = (t2 >> 9) & 3;
    int pc = (t2 >> 8) & 1;
    int l  = (t2 >> 3) & 31;
    int j  = t2 & 7;
    int n = nb * 16 + (l & 15);
    int k = kb * 32 + pc * 16 + ((l >> 4) & 1) * 8 + j;
    wsW[tid] = (_Float16)W2[k * DGLOB + n];
  }
}

// -------------------------- kernel 2: fused edge MLP ------------------------
__global__ __launch_bounds__(256, 1) void mp_edge_mlp(
    const float* __restrict__ xs, const float* __restrict__ xt,
    const int* __restrict__ ei, const float* __restrict__ ea,
    const float* __restrict__ xu,
    const float* __restrict__ b1, const float* __restrict__ b2,
    const float* __restrict__ gamma,
    const _Float16* __restrict__ wsW,
    float* __restrict__ out, int E, int tiles) {
  extern __shared__ char smem[];
  _Float16* sW1 = (_Float16*)smem;          // 65536 halves
  _Float16* sW2 = sW1 + 65536;              // 16384 halves
  _Float16* sH  = sW2 + 16384;              // 8 waves * 4096 halves
  float* sB1 = (float*)(sH + 8 * 4096);     // 256
  float* sB2 = sB1 + 256;                   // 64
  float* sG  = sB2 + 64;                    // 64

  // ---- stage packed f16 weights ws -> LDS (163840 B = 10240 x uint4) ----
  {
    const uint4* gsrc = (const uint4*)wsW;
    uint4* ldst = (uint4*)sW1;
    for (int i = threadIdx.x; i < 10240; i += 256) ldst[i] = gsrc[i];
    if (threadIdx.x < 256) sB1[threadIdx.x] = b1[threadIdx.x];
    if (threadIdx.x < 64) {
      sB2[threadIdx.x] = b2[threadIdx.x];
      sG[threadIdx.x]  = gamma[threadIdx.x];
    }
  }
  __syncthreads();

  const int lane = threadIdx.x & 31;
  const int wave = threadIdx.x >> 5;
  const int lm   = lane & 15;
  const bool hiHalf = lane >= 16;
  const int hOff = hiHalf ? 8 : 0;
  _Float16* myH = sH + wave * 4096;   // 8 kb-tiles * 512 halves

  for (int t = blockIdx.x * 8 + wave; t < tiles; t += gridDim.x * 8) {
    int e  = t * 16 + lm;
    int ec = e < E ? e : (E - 1);
    int src = ei[ec];
    int tgt = ei[E + ec];

    // concat segment base pointers (K-blocks kb=2s, 2s+1 live in segment s)
    const float* segp[4];
    segp[0] = xs + (size_t)src * DGLOB;
    segp[1] = xt + (size_t)tgt * DGLOB;
    segp[2] = ea + (size_t)ec * DGLOB;
    segp[3] = xu;

    // ---------------- layer 1: [16x256] @ W1[256x256], K-outer -------------
    v8f acc[16];
#pragma unroll
    for (int nb = 0; nb < 16; ++nb)
#pragma unroll
      for (int i = 0; i < 8; ++i) acc[nb][i] = 0.0f;

    // manual 1-deep prefetch of the A-gather (4 x f4 per lane)
    // A-frag lane layout: lanes 0-15 -> K {b..b+7, b+16..b+23}
    //                     lanes16-31 -> K {b+8..b+15, b+24..b+31}
    f4 c0, c1, c2, c3;
    {
      const float* rp = segp[0] + hOff;          // kb=0: k0 = hOff
      c0 = *(const f4*)(rp);
      c1 = *(const f4*)(rp + 4);
      c2 = *(const f4*)(rp + 16);
      c3 = *(const f4*)(rp + 20);
    }

#pragma unroll
    for (int kb = 0; kb < 8; ++kb) {
      h16 af = cat16(pk8(c0, c1), pk8(c2, c3));

      if (kb < 7) {  // issue next K-block's gather before the WMMA burst
        const float* rp = segp[(kb + 1) >> 1] + (((kb + 1) & 1) << 5) + hOff;
        c0 = *(const f4*)(rp);
        c1 = *(const f4*)(rp + 4);
        c2 = *(const f4*)(rp + 16);
        c3 = *(const f4*)(rp + 20);
      }

      // fragment-packed B tiles: conflict-free 16B lane stride,
      // pipelined TWO tiles deep (wait dscnt<=2 before each WMMA)
      const _Float16* bkb = sW1 + kb * 8192 + lane * 8;
      h16 bf0 = ldfrag(bkb);
      h16 bf1 = ldfrag(bkb + 512);
#pragma unroll
      for (int nb = 0; nb < 16; ++nb) {
        h16 bf2 = bf1;
        if (nb < 14) bf2 = ldfrag(bkb + (nb + 2) * 512);
        acc[nb] = __builtin_amdgcn_wmma_f32_16x16x32_f16(
            false, af, false, bf0, (short)0, acc[nb], false, false);
        bf0 = bf1;
        bf1 = bf2;
      }
      __builtin_amdgcn_sched_barrier(0);  // cap prefetch depth at 1 K-block
    }

    // bias + LeakyReLU, scatter H into A-fragment-packed layout:
    // H elem (row m, col n): kb2=n>>5, q=(n>>4)&1, l=m+((n&15)>=8?16:0),
    // j=n&7  ->  addr = kb2*512 + q*256 + l*8 + j
#pragma unroll
    for (int nb = 0; nb < 16; ++nb) {
      float bias = sB1[nb * 16 + lm];
      int lbase = (lm >= 8) ? 16 : 0;
      _Float16* hp = myH + (nb >> 1) * 512 + (nb & 1) * 256 +
                     (lbase + hOff) * 8 + (lm & 7);
#pragma unroll
      for (int r = 0; r < 8; ++r) {   // C/D: VGPR r -> row r (lo) / r+8 (hi)
        float v = acc[nb][r] + bias;
        v = (v >= 0.0f) ? v : LEAKY * v;
        hp[r * 8] = (_Float16)v;
      }
    }

    // ---------------- layer 2: H[16x256] @ W2[256x64], K-outer -------------
    v8f acc2[4];
#pragma unroll
    for (int nb = 0; nb < 4; ++nb)
#pragma unroll
      for (int i = 0; i < 8; ++i) acc2[nb][i] = 0.0f;

#pragma unroll
    for (int kb = 0; kb < 8; ++kb) {
      h16 a2 = ldfrag(myH + kb * 512 + lane * 8);   // packed A tile

      const _Float16* bkb = sW2 + kb * 2048 + lane * 8;
      h16 bf0 = ldfrag(bkb);
      h16 bf1 = ldfrag(bkb + 512);
#pragma unroll
      for (int nb = 0; nb < 4; ++nb) {
        h16 bf2 = bf1;
        if (nb < 2) bf2 = ldfrag(bkb + (nb + 2) * 512);
        acc2[nb] = __builtin_amdgcn_wmma_f32_16x16x32_f16(
            false, a2, false, bf0, (short)0, acc2[nb], false, false);
        bf0 = bf1;
        bf1 = bf2;
      }
    }

    // ---------------- bias2 + RMSNorm + store ------------------------------
    float vout[4][8];
#pragma unroll
    for (int nb = 0; nb < 4; ++nb) {
      float bias = sB2[nb * 16 + lm];
#pragma unroll
      for (int r = 0; r < 8; ++r) vout[nb][r] = acc2[nb][r] + bias;
    }

#pragma unroll
    for (int r = 0; r < 8; ++r) {
      float s = vout[0][r] * vout[0][r] + vout[1][r] * vout[1][r] +
                vout[2][r] * vout[2][r] + vout[3][r] * vout[3][r];
      // row's 64 cols live across the 16 lanes of this half-wave
      s += __shfl_xor(s, 1, 16);
      s += __shfl_xor(s, 2, 16);
      s += __shfl_xor(s, 4, 16);
      s += __shfl_xor(s, 8, 16);
      float inv = rsqrtf(s * (1.0f / 64.0f) + RMSEPS);
      int row = t * 16 + r + hOff;
      if (row < E) {
        float* orow = out + (size_t)row * DGLOB;
#pragma unroll
        for (int nb = 0; nb < 4; ++nb)
          orow[nb * 16 + lm] = vout[nb][r] * inv * sG[nb * 16 + lm];
      }
    }
  }
}

// -------------------------- host launch -------------------------------------
extern "C" void kernel_launch(void* const* d_in, const int* in_sizes, int n_in,
                              void* d_out, int out_size, void* d_ws, size_t ws_size,
                              hipStream_t stream) {
  const float* xs    = (const float*)d_in[0];
  const float* xt    = (const float*)d_in[1];
  const int*   ei    = (const int*)d_in[2];
  const float* ea    = (const float*)d_in[3];
  const float* xu    = (const float*)d_in[4];
  const float* W1    = (const float*)d_in[5];
  const float* b1    = (const float*)d_in[6];
  const float* W2    = (const float*)d_in[7];
  const float* b2    = (const float*)d_in[8];
  const float* gamma = (const float*)d_in[9];
  float* out = (float*)d_out;

  int E = in_sizes[3] / DGLOB;          // edge_attr is [E, 64]
  int tiles = (E + 15) / 16;
  _Float16* wsW = (_Float16*)d_ws;      // needs 81920 * 2 = 163840 bytes

  // one-time (per call, deterministic) f32->f16 fragment-packed weight prep
  mp_pack_w<<<320, 256, 0, stream>>>(W1, W2, wsW);

  // allow >64KB dynamic LDS (no-op if unsupported; ignore error)
  (void)hipFuncSetAttribute((const void*)mp_edge_mlp,
                            hipFuncAttributeMaxDynamicSharedMemorySize,
                            SMEM_BYTES);

  mp_edge_mlp<<<512, 256, SMEM_BYTES, stream>>>(
      xs, xt, ei, ea, xu, b1, b2, gamma, wsW, out, E, tiles);
}
